// ShiftInvariantAttention_84112639525635
// MI455X (gfx1250) — matrix operand order repaired and checked by reading, not compiled, not run on hardware
//
#include <hip/hip_runtime.h>

#define DEV __device__ __forceinline__

typedef __attribute__((ext_vector_type(16))) __bf16 v16bf;
typedef __attribute__((ext_vector_type(8)))  float  v8f;

struct U4 { unsigned int x, y, z, w; };

// Build a 16x32 (A) or 32x16 (B) bf16 WMMA fragment from two 16-byte chunks.
DEV v16bf load_frag(const __bf16* p0, const __bf16* p1) {
  union { v16bf v; U4 u[2]; } f;
  f.u[0] = *reinterpret_cast<const U4*>(p0);
  f.u[1] = *reinterpret_cast<const U4*>(p1);
  return f.v;
}

DEV v8f wmma_bf16(v16bf a, v16bf b, v8f c) {
  // (neg_a, A, neg_b, B, c_mod, C, reuse_a, reuse_b)
  return __builtin_amdgcn_wmma_f32_16x16x32_bf16(false, a, false, b, (short)0, c,
                                                 false, false);
}

namespace cfg {
constexpr int L  = 1024;           // sequence length
constexpr int D  = 512;            // model dim
constexpr int H  = 8;              // heads
constexpr int DH = 64;             // head dim
constexpr int NB = 8;              // N*S flattened batch
constexpr int R  = NB * L;         // 8192 rows
constexpr int D2 = 2 * D;          // 1024
constexpr int BH = NB * H;         // 64
constexpr int P  = 3;
constexpr float EPS = 1e-5f;
}

// ---------------------------------------------------------------- fp32->bf16
__global__ void k_cvt(const float* __restrict__ s, __bf16* __restrict__ d, int n) {
  for (int i = blockIdx.x * blockDim.x + threadIdx.x; i < n;
       i += gridDim.x * blockDim.x)
    d[i] = (__bf16)s[i];
}

// ----------------------------------------------- generic bf16 GEMM: X @ W^T
// A: [M,K] bf16 row-major. W: [Nc,K] bf16 row-major. One wave -> 16x64 tile.
__global__ void k_gemm(const __bf16* __restrict__ A, const __bf16* __restrict__ W,
                       const float* __restrict__ bias, float* __restrict__ outF,
                       __bf16* __restrict__ outB, int M, int Nc, int K, int relu) {
  const int wv   = (blockIdx.x << 3) + (threadIdx.x >> 5);
  const int lane = threadIdx.x & 31;
  const int tilesN = Nc >> 6;
  const int tm = wv / tilesN, tn = wv % tilesN;
  if (tm * 16 >= M) return;
  const int hl = lane >> 4, lm = lane & 15;

  const __bf16* arow = A + (size_t)(tm * 16 + lm) * K;
  const __bf16* w0 = W + (size_t)(tn * 64 + lm) * K;
  const __bf16* w1 = w0 + (size_t)16 * K;
  const __bf16* w2 = w0 + (size_t)32 * K;
  const __bf16* w3 = w0 + (size_t)48 * K;

  v8f acc[4] = {};
  for (int k0 = 0; k0 < K; k0 += 32) {
    v16bf a = load_frag(arow + k0 + hl * 8, arow + k0 + 16 + hl * 8);
    acc[0] = wmma_bf16(a, load_frag(w0 + k0 + hl * 8, w0 + k0 + 16 + hl * 8), acc[0]);
    acc[1] = wmma_bf16(a, load_frag(w1 + k0 + hl * 8, w1 + k0 + 16 + hl * 8), acc[1]);
    acc[2] = wmma_bf16(a, load_frag(w2 + k0 + hl * 8, w2 + k0 + 16 + hl * 8), acc[2]);
    acc[3] = wmma_bf16(a, load_frag(w3 + k0 + hl * 8, w3 + k0 + 16 + hl * 8), acc[3]);
  }
#pragma unroll
  for (int r = 0; r < 8; ++r) {
    const size_t row = (size_t)(tm * 16 + hl * 8 + r);
#pragma unroll
    for (int t = 0; t < 4; ++t) {
      const int col = tn * 64 + t * 16 + lm;
      float v = acc[t][r] + bias[col];
      if (relu) v = fmaxf(v, 0.f);
      if (outF) outF[row * Nc + col] = v;
      else      outB[row * Nc + col] = (__bf16)v;
    }
  }
}

// --------------------------------- build attention operands (bf16, fused trig)
// Abuf[bh][l][0:64]=q*sin(pq)  [64:128]=q*cos(pq)
// Bbuf[bh][l][0:64]=k*cos(pk-b) [64:128]=-k*sin(pk-b)   (pk-b = Pk@Wp^T)
// Vt  [bh][c][l] = v  (transposed for contiguous K loads in P@V)
__global__ void k_prep(const float* __restrict__ qf, const float* __restrict__ kf,
                       const float* __restrict__ vf, const float* __restrict__ Pq,
                       const float* __restrict__ Pk, const float* __restrict__ Wp,
                       const float* __restrict__ bp, __bf16* __restrict__ Ab,
                       __bf16* __restrict__ Bb, __bf16* __restrict__ Vt) {
  using namespace cfg;
  const int e = blockIdx.x * blockDim.x + threadIdx.x;   // [0, R*D)
  const int row = e / D, d = e % D;
  const int b = row / L, l = row % L, h = d / DH, c = d % DH;
  const int bh = b * H + h;

  const float w0 = Wp[d * P + 0], w1 = Wp[d * P + 1], w2 = Wp[d * P + 2];
  const float pq = Pq[row * P + 0] * w0 + Pq[row * P + 1] * w1 +
                   Pq[row * P + 2] * w2 + bp[d];
  const float pk = Pk[row * P + 0] * w0 + Pk[row * P + 1] * w1 +
                   Pk[row * P + 2] * w2;                 // pk - b
  float sq, cq, sk, ck;
  __sincosf(pq, &sq, &cq);
  __sincosf(pk, &sk, &ck);

  const float q = qf[e], k = kf[e], v = vf[e];
  const size_t ar = ((size_t)bh * L + l) * 128;
  Ab[ar + c]      = (__bf16)(q * sq);
  Ab[ar + 64 + c] = (__bf16)(q * cq);
  Bb[ar + c]      = (__bf16)(k * ck);
  Bb[ar + 64 + c] = (__bf16)(-k * sk);
  Vt[((size_t)bh * DH + c) * L + l] = (__bf16)v;
}

// ---------------------------------------------- flash attention, wave per 16 q
__global__ void k_attn(const __bf16* __restrict__ Ab, const __bf16* __restrict__ Bb,
                       const __bf16* __restrict__ Vt, float* __restrict__ T) {
  using namespace cfg;
  __shared__ __bf16 smem[8][512];                       // 16x32 P tile per wave
  const int w = threadIdx.x >> 5, lane = threadIdx.x & 31;
  const int hl = lane >> 4, lm = lane & 15;
  const int bh = blockIdx.x >> 3, qb = blockIdx.x & 7;
  const int q0 = qb * 128 + w * 16;

  const __bf16* arow = Ab + ((size_t)bh * L + q0 + lm) * 128;
  v16bf afrag[4];
#pragma unroll
  for (int t = 0; t < 4; ++t)
    afrag[t] = load_frag(arow + t * 32 + hl * 8, arow + t * 32 + 16 + hl * 8);

  float rmax[8], rsum[8];
#pragma unroll
  for (int r = 0; r < 8; ++r) { rmax[r] = -1e30f; rsum[r] = 0.f; }
  v8f o[4] = {};

  for (int j0 = 0; j0 < L; j0 += 32) {
    const __bf16* b0 = Bb + ((size_t)bh * L + j0 + lm) * 128;
    const __bf16* b1 = b0 + (size_t)16 * 128;
    v8f sa = {}, sb = {};
#pragma unroll
    for (int t = 0; t < 4; ++t) {
      sa = wmma_bf16(afrag[t], load_frag(b0 + t * 32 + hl * 8,
                                         b0 + t * 32 + 16 + hl * 8), sa);
      sb = wmma_bf16(afrag[t], load_frag(b1 + t * 32 + hl * 8,
                                         b1 + t * 32 + 16 + hl * 8), sb);
    }
#pragma unroll
    for (int r = 0; r < 8; ++r) {
      float tm = fmaxf(sa[r], sb[r]);
      tm = fmaxf(tm, __shfl_xor(tm, 1, 32));
      tm = fmaxf(tm, __shfl_xor(tm, 2, 32));
      tm = fmaxf(tm, __shfl_xor(tm, 4, 32));
      tm = fmaxf(tm, __shfl_xor(tm, 8, 32));            // row max over 16 lanes
      const float mnew  = fmaxf(rmax[r], tm);
      const float scale = __expf(rmax[r] - mnew);
      const float ea = __expf(sa[r] - mnew);
      const float eb = __expf(sb[r] - mnew);
      float ts = ea + eb;
      ts += __shfl_xor(ts, 1, 32);
      ts += __shfl_xor(ts, 2, 32);
      ts += __shfl_xor(ts, 4, 32);
      ts += __shfl_xor(ts, 8, 32);
      rsum[r] = rsum[r] * scale + ts;
      rmax[r] = mnew;
      o[0][r] *= scale; o[1][r] *= scale; o[2][r] *= scale; o[3][r] *= scale;
      const int m = hl * 8 + r;                         // C-layout -> LDS 16x32
      smem[w][m * 32 + lm]      = (__bf16)ea;
      smem[w][m * 32 + 16 + lm] = (__bf16)eb;
    }
    asm volatile("s_wait_dscnt 0" ::: "memory");        // wave-local LDS RAW
    v16bf pf = load_frag(&smem[w][lm * 32 + hl * 8],
                         &smem[w][lm * 32 + 16 + hl * 8]);
    const __bf16* vr = Vt + ((size_t)bh * DH + lm) * L + j0 + hl * 8;
#pragma unroll
    for (int t = 0; t < 4; ++t)
      o[t] = wmma_bf16(pf, load_frag(vr + (size_t)t * 16 * L,
                                     vr + (size_t)t * 16 * L + 16), o[t]);
    asm volatile("s_wait_dscnt 0" ::: "memory");        // before next overwrite
  }

  const int b = bh >> 3, h = bh & 7;
#pragma unroll
  for (int r = 0; r < 8; ++r) {
    const float inv = 1.f / rsum[r];
    const size_t row = (size_t)b * L + q0 + hl * 8 + r;
#pragma unroll
    for (int t = 0; t < 4; ++t)
      T[row * D + h * DH + t * 16 + lm] = o[t][r] * inv;
  }
}

// ------------------------------------------------------------------ batchnorm
__global__ void k_bn_zero(float* sums) {
  const int t = threadIdx.x;
  sums[t] = 0.f; sums[t + 256] = 0.f; sums[t + 512] = 0.f; sums[t + 768] = 0.f;
}

__global__ void k_bn_stats(const float* __restrict__ U, float* __restrict__ sums) {
  using namespace cfg;
  const int col = threadIdx.x;                 // handles col and col+256
  const int base = blockIdx.x * 128;
  float s0 = 0.f, q0 = 0.f, s1 = 0.f, q1 = 0.f;
  for (int r = 0; r < 128; ++r) {
    const float x = U[(size_t)(base + r) * D + col];
    const float y = U[(size_t)(base + r) * D + col + 256];
    s0 += x; q0 += x * x; s1 += y; q1 += y * y;
  }
  atomicAdd(&sums[col], s0);
  atomicAdd(&sums[col + 256], s1);
  atomicAdd(&sums[512 + col], q0);
  atomicAdd(&sums[512 + col + 256], q1);
}

__global__ void k_bn_apply(const float* __restrict__ U, const float* __restrict__ Q,
                           const float* __restrict__ sums,
                           const float* __restrict__ gamma,
                           const float* __restrict__ beta, float* __restrict__ out) {
  using namespace cfg;
  const int e = blockIdx.x * blockDim.x + threadIdx.x;  // [0, R*D)
  const int c = e % D;
  const float mu  = sums[c] * (1.f / (float)R);
  const float var = sums[512 + c] * (1.f / (float)R) - mu * mu;
  out[e] = Q[e] + gamma[c] * (U[e] - mu) * rsqrtf(var + EPS) + beta[c];
}

// ------------------------------------------------------------------- launcher
extern "C" void kernel_launch(void* const* d_in, const int* in_sizes, int n_in,
                              void* d_out, int out_size, void* d_ws, size_t ws_size,
                              hipStream_t stream) {
  using namespace cfg;
  const float* Q  = (const float*)d_in[0];
  const float* K  = (const float*)d_in[1];
  const float* Pq = (const float*)d_in[2];
  const float* Pk = (const float*)d_in[3];
  const float* Wq = (const float*)d_in[4];
  const float* bq = (const float*)d_in[5];
  const float* Wk = (const float*)d_in[6];
  const float* bk = (const float*)d_in[7];
  const float* Wv = (const float*)d_in[8];
  const float* bv = (const float*)d_in[9];
  const float* Wp = (const float*)d_in[10];
  const float* bp = (const float*)d_in[11];
  const float* We = (const float*)d_in[12];
  const float* be = (const float*)d_in[13];
  const float* Wc = (const float*)d_in[14];
  const float* bc = (const float*)d_in[15];
  const float* gamma = (const float*)d_in[16];
  const float* beta  = (const float*)d_in[17];
  float* out = (float*)d_out;

  // workspace layout (~113 MB), 256B-aligned regions with explicit reuse
  char* ws = (char*)d_ws;
  size_t off = 0;
  auto alloc = [&](size_t bytes) -> char* {
    char* p = ws + off;
    off += (bytes + 255) & ~(size_t)255;
    return p;
  };
  __bf16* Wqb = (__bf16*)alloc((size_t)D * D * 2);
  __bf16* Wkb = (__bf16*)alloc((size_t)D * D * 2);
  __bf16* Wvb = (__bf16*)alloc((size_t)D * D * 2);
  __bf16* Web = (__bf16*)alloc((size_t)D2 * D * 2);
  __bf16* Wcb = (__bf16*)alloc((size_t)D * D2 * 2);
  __bf16* Qb  = (__bf16*)alloc((size_t)R * D * 2);   // dead after v-GEMM
  __bf16* Kb  = (__bf16*)alloc((size_t)R * D * 2);   // (contiguous with Qb)
  float*  qf  = (float*)alloc((size_t)R * D * 4);    // dead after prep
  float*  kf  = (float*)alloc((size_t)R * D * 4);
  float*  vf  = (float*)alloc((size_t)R * D * 4);
  __bf16* Abuf = (__bf16*)alloc((size_t)BH * L * 128 * 2);
  __bf16* Bbuf = (__bf16*)alloc((size_t)BH * L * 128 * 2);
  __bf16* Vt   = (__bf16*)alloc((size_t)BH * DH * L * 2);
  float*  sums = (float*)alloc(1024 * 4);
  // aliases over dead regions
  float*  Tbuf = qf;                 // attention output (R x D fp32)
  __bf16* Tbf  = (__bf16*)kf;        // bf16 copy for FFN
  __bf16* Ebuf = (__bf16*)vf;        // FFN hidden (R x 2D bf16)
  float*  Ubuf = (float*)Qb;         // FFN output  (R x D fp32, spans Qb+Kb)

  // 1) precision conversion
  k_cvt<<<512,  256, 0, stream>>>(Wq, Wqb, D * D);
  k_cvt<<<512,  256, 0, stream>>>(Wk, Wkb, D * D);
  k_cvt<<<512,  256, 0, stream>>>(Wv, Wvb, D * D);
  k_cvt<<<1024, 256, 0, stream>>>(We, Web, D2 * D);
  k_cvt<<<1024, 256, 0, stream>>>(Wc, Wcb, D * D2);
  k_cvt<<<2048, 256, 0, stream>>>(Q, Qb, R * D);
  k_cvt<<<2048, 256, 0, stream>>>(K, Kb, R * D);

  // 2) q/k/v projections (WMMA bf16, fp32 accumulate)
  k_gemm<<<512, 256, 0, stream>>>(Qb, Wqb, bq, qf, nullptr, R, D, D, 0);
  k_gemm<<<512, 256, 0, stream>>>(Kb, Wkb, bk, kf, nullptr, R, D, D, 0);
  k_gemm<<<512, 256, 0, stream>>>(Kb, Wvb, bv, vf, nullptr, R, D, D, 0);

  // 3) trig-fused operand build
  k_prep<<<R * D / 256, 256, 0, stream>>>(qf, kf, vf, Pq, Pk, Wp, bp,
                                          Abuf, Bbuf, Vt);

  // 4) flash attention (online softmax, score matrix stays on-chip)
  k_attn<<<BH * (L / 128), 256, 0, stream>>>(Abuf, Bbuf, Vt, Tbuf);

  // 5) FFN
  k_cvt<<<2048, 256, 0, stream>>>(Tbuf, Tbf, R * D);
  k_gemm<<<1024, 256, 0, stream>>>(Tbf, Web, be, nullptr, Ebuf, R, D2, D, 1);
  k_gemm<<<512,  256, 0, stream>>>(Ebuf, Wcb, bc, Ubuf, nullptr, R, D, D2, 0);

  // 6) batchnorm (training stats) + residual
  k_bn_zero<<<1, 256, 0, stream>>>(sums);
  k_bn_stats<<<R / 128, 256, 0, stream>>>(Ubuf, sums);
  k_bn_apply<<<R * D / 256, 256, 0, stream>>>(Ubuf, Q, sums, gamma, beta, out);
}
